// SelfAttention_53051436040559
// MI455X (gfx1250) — matrix operand order repaired and checked by reading, not compiled, not run on hardware
//
#include <hip/hip_runtime.h>

// ---------------------------------------------------------------------------
// Problem constants (reference: B=4, S=2048, D=1024, H=16, HD=64)
// ---------------------------------------------------------------------------
constexpr int Bc = 4;
constexpr int Sc = 2048;
constexpr int Dc = 1024;
constexpr int Hc = 16;
constexpr int HDc = 64;

typedef __attribute__((ext_vector_type(16))) __bf16 v16bf;
typedef __attribute__((ext_vector_type(8)))  float  v8f;
typedef __attribute__((ext_vector_type(4)))  float  v4f;
typedef __attribute__((ext_vector_type(4)))  int    v4i;

union FragBF  { v16bf v; v4f f4[2]; };
union Pack128 { v4f f4; unsigned short us[8]; };

// ---------------------------------------------------------------------------
// CDNA5 async global->LDS path (ASYNCcnt-tracked, no VGPR transit), with a
// synchronous fallback so the source compiles on toolchains without the
// builtin. Builtin signature (probed via diagnostics): takes v4i pointers in
// AS1 (global) and AS3 (LDS). Pointers are built via integer casts:
// generic-pointer low 32 bits == LDS byte offset (ISA 10.2 aperture rules).
// ---------------------------------------------------------------------------
#if defined(__HIP_DEVICE_COMPILE__) && \
    __has_builtin(__builtin_amdgcn_global_load_async_to_lds_b128)
#define HAS_ASYNC_LDS 1
#else
#define HAS_ASYNC_LDS 0
#endif

typedef __attribute__((address_space(1))) v4i glb_v4i;
typedef __attribute__((address_space(3))) v4i lds_v4i;

// copy 16B (8 bf16) global -> LDS, per lane
__device__ __forceinline__ void async_cp16(unsigned short* lds,
                                           const unsigned short* g) {
#if HAS_ASYNC_LDS
  glb_v4i* gp = (glb_v4i*)(unsigned long long)g;
  lds_v4i* lp = (lds_v4i*)(unsigned long long)(unsigned)(unsigned long long)lds;
  __builtin_amdgcn_global_load_async_to_lds_b128(gp, lp, 0, 0);
#else
  *(v4f*)lds = *(const v4f*)g;
#endif
}

__device__ __forceinline__ void wait_async0() {
#if HAS_ASYNC_LDS
#if __has_builtin(__builtin_amdgcn_s_wait_asynccnt)
  __builtin_amdgcn_s_wait_asynccnt(0);
#else
  asm volatile("s_wait_asynccnt 0x0" ::: "memory");
#endif
#endif
}

// round-to-nearest-even f32 -> bf16 (bit pattern in ushort)
__device__ __forceinline__ unsigned short f2bf(float f) {
  union { float f; unsigned u; } c; c.f = f;
  unsigned u = c.u + 0x7FFFu + ((c.u >> 16) & 1u);
  return (unsigned short)(u >> 16);
}

// ---------------------------------------------------------------------------
// WMMA fragment loads from LDS (bf16, wave32 layouts per CDNA5 ISA 7.12.2)
// A (16xK=32): lanes 0-15 row M=l, K {0..7,16..23}; lanes 16-31 K {8..15,24..31}
// B (K=32x16): lanes 0-15 col N=l, K {0..7,16..23}; lanes 16-31 K {8..15,24..31}
// stride in bf16 elements; all chunk offsets 16B aligned (stride % 8 == 0)
// ---------------------------------------------------------------------------
__device__ __forceinline__ v16bf ld_afrag(const unsigned short* s, int row0,
                                          int ks, int stride) {
  const int l = threadIdx.x & 31;
  const int r = row0 + (l & 15);
  const int kh = (l >> 4) * 8;
  FragBF f;
  f.f4[0] = *(const v4f*)(s + r * stride + ks + kh);
  f.f4[1] = *(const v4f*)(s + r * stride + ks + kh + 16);
  return f.v;
}

__device__ __forceinline__ v16bf ld_bfrag(const unsigned short* s, int col0,
                                          int ks, int stride) {
  const int l = threadIdx.x & 31;
  const int c = col0 + (l & 15);
  const int kh = (l >> 4) * 8;
  FragBF f;
  f.f4[0] = *(const v4f*)(s + c * stride + ks + kh);
  f.f4[1] = *(const v4f*)(s + c * stride + ks + kh + 16);
  return f.v;
}

// ---------------------------------------------------------------------------
// f32 -> bf16 bulk convert
// ---------------------------------------------------------------------------
__global__ void cvt_f32_bf16(const float* __restrict__ in,
                             unsigned short* __restrict__ out, int n) {
  int i = blockIdx.x * blockDim.x + threadIdx.x;
  const int stride = gridDim.x * blockDim.x;
  for (; i < n; i += stride) out[i] = f2bf(in[i]);
}

// ---------------------------------------------------------------------------
// Tiled WMMA GEMM:  Y[m,n] = sum_k A[m,k] * W[n,k] + bias[n]
//   A: [8192,1024] bf16 row-major, W: [1024,1024] bf16 row-major ([n,k])
//   block tile 128x128, 8 waves (4 along M x 2 along N), wave tile 32x64
//   double-buffered 32-deep k-stages filled by async global->LDS copies
//   mode 0: write bf16 to [B,H,S,HD]    (QKV projections)
//   mode 1: bias + LeakyReLU + row mask, write f32 [B,S,D]  (out projection)
// ---------------------------------------------------------------------------
__global__ __launch_bounds__(256) void gemm_bf16(
    const unsigned short* __restrict__ A, const unsigned short* __restrict__ W,
    const float* __restrict__ bias, const float* __restrict__ rowmask,
    unsigned short* __restrict__ outBF, float* __restrict__ outF, int mode) {
  constexpr int K = Dc;
  constexpr int LP = 40;  // LDS row stride (bf16): 80B, 16B-aligned chunks
  __shared__ unsigned short sA[2][128 * LP];
  __shared__ unsigned short sB[2][128 * LP];

  const int t = threadIdx.x;
  const int lane = t & 31, wave = t >> 5;
  const int waveM = wave >> 1, waveN = wave & 1;
  const int m0 = blockIdx.y * 128, n0 = blockIdx.x * 128;

  // per-thread staging coords: 128 rows x 4 chunks(8 elems) = 512 copies,
  // 2 per thread per matrix
  const int r0 = t >> 2, c0 = (t & 3) * 8;
  const int r1 = (t + 256) >> 2, c1 = ((t + 256) & 3) * 8;

  auto stage = [&](int buf, int k0) {
    async_cp16(&sA[buf][r0 * LP + c0], A + (size_t)(m0 + r0) * K + k0 + c0);
    async_cp16(&sA[buf][r1 * LP + c1], A + (size_t)(m0 + r1) * K + k0 + c1);
    async_cp16(&sB[buf][r0 * LP + c0], W + (size_t)(n0 + r0) * K + k0 + c0);
    async_cp16(&sB[buf][r1 * LP + c1], W + (size_t)(n0 + r1) * K + k0 + c1);
  };

  v8f acc[2][4];
  for (int i = 0; i < 2; i++)
    for (int j = 0; j < 4; j++)
      for (int e = 0; e < 8; e++) acc[i][j][e] = 0.f;

  stage(0, 0);
  wait_async0();
  __syncthreads();

  int cur = 0;
  for (int k0 = 0; k0 < K; k0 += 32) {
    if (k0 + 32 < K) stage(cur ^ 1, k0 + 32);  // prefetch next stage (async)

    v16bf af[2], bf[4];
    for (int mt = 0; mt < 2; mt++)
      af[mt] = ld_afrag(sA[cur], waveM * 32 + mt * 16, 0, LP);
    for (int nt = 0; nt < 4; nt++)
      bf[nt] = ld_bfrag(sB[cur], waveN * 64 + nt * 16, 0, LP);
    for (int mt = 0; mt < 2; mt++)
      for (int nt = 0; nt < 4; nt++)
        acc[mt][nt] = __builtin_amdgcn_wmma_f32_16x16x32_bf16(
            false, af[mt], false, bf[nt], (short)0, acc[mt][nt], false, false);

    wait_async0();   // my async fills of the other buffer have landed
    __syncthreads(); // everyone's landed; everyone done reading cur
    cur ^= 1;
  }

  // epilogue: C/D layout -> VGPR j: M=j (lanes0-15) / j+8 (lanes16-31), N=lane&15
  for (int mt = 0; mt < 2; mt++)
    for (int nt = 0; nt < 4; nt++)
      for (int j = 0; j < 8; j++) {
        const int m = m0 + waveM * 32 + mt * 16 + j + ((lane >> 4) << 3);
        const int n = n0 + waveN * 64 + nt * 16 + (lane & 15);
        float y = acc[mt][nt][j] + bias[n];
        if (mode == 0) {
          const int b = m >> 11, s = m & (Sc - 1);
          const int h = n >> 6, hd = n & 63;
          outBF[(((size_t)b * Hc + h) * Sc + s) * HDc + hd] = f2bf(y);
        } else {
          y = (y >= 0.f) ? y : 0.01f * y;
          y *= rowmask[m];
          outF[(size_t)m * Dc + n] = y;
        }
      }
}

// ---------------------------------------------------------------------------
// Flash attention: one block per (b*H+h, 128 q rows); 8 waves x 16 q rows.
// Double-buffered K tile [64,64] (async fill) + transposed V tile [64(d),64(k)]
// (sync VGPR transpose). Online softmax across S in 64-key blocks; P bounced
// through per-wave LDS to convert score D-layout into WMMA A-layout.
// ---------------------------------------------------------------------------
__global__ __launch_bounds__(256) void flash_attn(
    const unsigned short* __restrict__ Qb, const unsigned short* __restrict__ Kb,
    const unsigned short* __restrict__ Vb, const float* __restrict__ mask,
    float* __restrict__ ctx) {
  constexpr int LP = 72;
  __shared__ unsigned short sK[2][64 * LP];   // [krow][d]
  __shared__ unsigned short sV[2][64 * LP];   // transposed: [d][krow]
  __shared__ unsigned short sP[8 * 16 * LP];  // per-wave P tile [16][64]
  __shared__ float sMask[2][64];

  const int t = threadIdx.x, lane = t & 31, wave = t >> 5;
  const int bh = blockIdx.y;
  const int b = bh >> 4, h = bh & 15;
  const int q0 = blockIdx.x * 128 + wave * 16;
  const size_t qkvBase = (size_t)bh * Sc * HDc;

  // Q fragments (A-matrix, head dim 64 -> two 32-deep k steps), kept in regs
  v16bf qf[2];
  {
    const int qr = q0 + (lane & 15);
    const int kh = (lane >> 4) * 8;
    const unsigned short* qp = Qb + qkvBase + (size_t)qr * HDc;
    for (int ds = 0; ds < 2; ds++) {
      FragBF f;
      f.f4[0] = *(const v4f*)(qp + ds * 32 + kh);
      f.f4[1] = *(const v4f*)(qp + ds * 32 + kh + 16);
      qf[ds] = f.v;
    }
  }

  float mrun[8], lrun[8];
  v8f acc[4];
  for (int j = 0; j < 8; j++) { mrun[j] = -3.0e38f; lrun[j] = 0.f; }
  for (int nt = 0; nt < 4; nt++)
    for (int e = 0; e < 8; e++) acc[nt][e] = 0.f;

  unsigned short* pw = sP + wave * (16 * LP);

  auto stage = [&](int buf, int kb) {
    for (int i = 0; i < 2; i++) {
      const int idx = t + i * 256;
      const int row = idx >> 3, ch = (idx & 7) * 8;
      async_cp16(&sK[buf][row * LP + ch],
                 Kb + qkvBase + (size_t)(kb + row) * HDc + ch);
      Pack128 pv;
      pv.f4 = *(const v4f*)(Vb + qkvBase + (size_t)(kb + row) * HDc + ch);
      for (int e = 0; e < 8; e++) sV[buf][(ch + e) * LP + row] = pv.us[e];
    }
    if (t < 64)
      sMask[buf][t] = (1.0f - mask[(size_t)b * Sc + kb + t]) * -10000.0f;
  };

  stage(0, 0);
  wait_async0();
  __syncthreads();

  int cur = 0;
  for (int kb = 0; kb < Sc; kb += 64) {
    if (kb + 64 < Sc) stage(cur ^ 1, kb + 64);  // prefetch next K/V tiles

    // scores: S = Q K^T / sqrt(64) + add_mask   (4 x 16-wide key subtiles)
    v8f sc[4];
    for (int kt = 0; kt < 4; kt++) {
      for (int e = 0; e < 8; e++) sc[kt][e] = 0.f;
      for (int ds = 0; ds < 2; ds++) {
        v16bf kfr = ld_bfrag(sK[cur], kt * 16, ds * 32, LP);
        sc[kt] = __builtin_amdgcn_wmma_f32_16x16x32_bf16(
            false, qf[ds], false, kfr, (short)0, sc[kt], false, false);
      }
      const float cm = sMask[cur][kt * 16 + (lane & 15)];
      for (int j = 0; j < 8; j++) sc[kt][j] = sc[kt][j] * 0.125f + cm;
    }

    // online softmax per q row; xor-shuffles 1..8 stay inside each 16-lane
    // half, matching the D-layout row striping
    for (int j = 0; j < 8; j++) {
      float rm = fmaxf(fmaxf(sc[0][j], sc[1][j]), fmaxf(sc[2][j], sc[3][j]));
      for (int m = 1; m < 16; m <<= 1) rm = fmaxf(rm, __shfl_xor(rm, m, 32));
      const float nm = fmaxf(mrun[j], rm);
      const float alpha = __expf(mrun[j] - nm);
      mrun[j] = nm;
      float ps = 0.f;
      for (int kt = 0; kt < 4; kt++) {
        const float p = __expf(sc[kt][j] - nm);
        sc[kt][j] = p;
        ps += p;
      }
      for (int m = 1; m < 16; m <<= 1) ps += __shfl_xor(ps, m, 32);
      lrun[j] = lrun[j] * alpha + ps;
      for (int nt = 0; nt < 4; nt++) acc[nt][j] *= alpha;
      // spill P row into per-wave LDS (row-major [q][k]) for A-fragment reload
      const int mr = j + ((lane >> 4) << 3);
      for (int kt = 0; kt < 4; kt++)
        pw[mr * LP + kt * 16 + (lane & 15)] = f2bf(sc[kt][j]);
    }

    // O += P @ V  (same-wave LDS store->load, DScnt keeps it ordered)
    for (int ks = 0; ks < 2; ks++) {
      v16bf pfr = ld_afrag(pw, 0, ks * 32, LP);
      for (int nt = 0; nt < 4; nt++) {
        v16bf vfr = ld_bfrag(sV[cur], nt * 16, ks * 32, LP);
        acc[nt] = __builtin_amdgcn_wmma_f32_16x16x32_bf16(
            false, pfr, false, vfr, (short)0, acc[nt], false, false);
      }
    }

    wait_async0();
    __syncthreads();
    cur ^= 1;
  }

  // normalize and scatter to ctx [B,S,D]
  for (int nt = 0; nt < 4; nt++)
    for (int j = 0; j < 8; j++) {
      const int q = q0 + j + ((lane >> 4) << 3);
      const int d = h * HDc + nt * 16 + (lane & 15);
      ctx[((size_t)b * Sc + q) * Dc + d] = acc[nt][j] / lrun[j];
    }
}

// ---------------------------------------------------------------------------
// residual add + LayerNorm over D=1024, one block per (b,s) row -> bf16
// ---------------------------------------------------------------------------
__global__ __launch_bounds__(256) void ln_fuse(
    const float* __restrict__ ctx, const float* __restrict__ resid,
    const float* __restrict__ g, const float* __restrict__ beta,
    unsigned short* __restrict__ xln) {
  __shared__ float rs[8], rs2[8], stats[2];
  const int t = threadIdx.x;
  const size_t base = (size_t)blockIdx.x * Dc;
  float x[4], s1 = 0.f, s2 = 0.f;
  for (int i = 0; i < 4; i++) {
    const int e = t + i * 256;
    x[i] = ctx[base + e] + resid[base + e];
    s1 += x[i];
    s2 += x[i] * x[i];
  }
  for (int m = 1; m < 32; m <<= 1) {
    s1 += __shfl_xor(s1, m, 32);
    s2 += __shfl_xor(s2, m, 32);
  }
  if ((t & 31) == 0) { rs[t >> 5] = s1; rs2[t >> 5] = s2; }
  __syncthreads();
  if (t == 0) {
    float a = 0.f, c = 0.f;
    for (int w = 0; w < 8; w++) { a += rs[w]; c += rs2[w]; }
    const float mu = a / Dc;
    const float var = c / Dc - mu * mu;
    stats[0] = mu;
    stats[1] = rsqrtf(var + 1e-5f);
  }
  __syncthreads();
  const float mu = stats[0], rstd = stats[1];
  for (int i = 0; i < 4; i++) {
    const int e = t + i * 256;
    xln[base + e] = f2bf((x[i] - mu) * rstd * g[e] + beta[e]);
  }
}

// ---------------------------------------------------------------------------
// launch
// ---------------------------------------------------------------------------
extern "C" void kernel_launch(void* const* d_in, const int* in_sizes, int n_in,
                              void* d_out, int out_size, void* d_ws,
                              size_t ws_size, hipStream_t stream) {
  (void)in_sizes; (void)n_in; (void)out_size; (void)ws_size;
  const float* inX  = (const float*)d_in[0];
  const float* mask = (const float*)d_in[1];
  const float* Wq = (const float*)d_in[2];  const float* bq = (const float*)d_in[3];
  const float* Wk = (const float*)d_in[4];  const float* bk = (const float*)d_in[5];
  const float* Wv = (const float*)d_in[6];  const float* bv = (const float*)d_in[7];
  const float* Wo = (const float*)d_in[8];  const float* bo = (const float*)d_in[9];
  const float* lng = (const float*)d_in[10];
  const float* lnb = (const float*)d_in[11];
  float* out = (float*)d_out;

  const size_t NX = (size_t)Bc * Sc * Dc;  // 8M activation elems
  const size_t NW = (size_t)Dc * Dc;       // 1M weight elems
  unsigned short* xb  = (unsigned short*)d_ws;
  unsigned short* wqb = xb + NX;
  unsigned short* wkb = wqb + NW;
  unsigned short* wvb = wkb + NW;
  unsigned short* wob = wvb + NW;
  unsigned short* Qb  = wob + NW;
  unsigned short* Kb  = Qb + NX;
  unsigned short* Vb  = Kb + NX;
  float* ctx = (float*)(Vb + NX);
  unsigned short* xln = (unsigned short*)(ctx + NX);

  cvt_f32_bf16<<<2048, 256, 0, stream>>>(inX, xb, (int)NX);
  cvt_f32_bf16<<<512, 256, 0, stream>>>(Wq, wqb, (int)NW);
  cvt_f32_bf16<<<512, 256, 0, stream>>>(Wk, wkb, (int)NW);
  cvt_f32_bf16<<<512, 256, 0, stream>>>(Wv, wvb, (int)NW);
  cvt_f32_bf16<<<512, 256, 0, stream>>>(Wo, wob, (int)NW);

  const dim3 gg(Dc / 128, (Bc * Sc) / 128);  // (8, 64)
  gemm_bf16<<<gg, 256, 0, stream>>>(xb, wqb, bq, nullptr, Qb, nullptr, 0);
  gemm_bf16<<<gg, 256, 0, stream>>>(xb, wkb, bk, nullptr, Kb, nullptr, 0);
  gemm_bf16<<<gg, 256, 0, stream>>>(xb, wvb, bv, nullptr, Vb, nullptr, 0);

  flash_attn<<<dim3(Sc / 128, Bc * Hc), 256, 0, stream>>>(Qb, Kb, Vb, mask, ctx);

  ln_fuse<<<Bc * Sc, 256, 0, stream>>>(ctx, inX, lng, lnb, xln);

  gemm_bf16<<<gg, 256, 0, stream>>>(xln, wob, bo, mask, nullptr, out, 1);
}
